// ResidualBlock_68564857913708
// MI455X (gfx1250) — matrix operand order repaired and checked by reading, not compiled, not run on hardware
//
#include <hip/hip_runtime.h>
#include <cstddef>
#include <cstdint>

// ---------------------------------------------------------------- types
typedef __attribute__((ext_vector_type(16))) __bf16 v16bf;
typedef __attribute__((ext_vector_type(8)))  float  v8f;

#define RMS_EPS 1e-5f
#define DSTATE  64
#define DCONV   4

__device__ __forceinline__ unsigned short f2bf(float f) {
  union { float f; unsigned u; } c; c.f = f;
  unsigned u = c.u;
  return (unsigned short)((u + 0x7fffu + ((u >> 16) & 1u)) >> 16);  // RNE
}
__device__ __forceinline__ float swishf(float v) { return v / (1.f + __expf(-v)); }

// ---------------------------------------------------------------- GEMM (bf16 WMMA, f32 accum)
// C[m,n] = sum_k A[m,k] * W[n,k] (+bias[n]) (optional swish) (+res[m,n])
// Block: 256 thr = 8 waves, tile 64x64, K step 32.
// Wave w: rows (w&3)*16..+16, cols (w>>2)*32..+32 -> two 16x16 WMMA tiles, A frag reused.
// Interior tiles use unconditional float4 loads (global_load_b128); only boundary
// tiles take the guarded scalar path.
#define BM 64
#define BN 64
#define BK 32
__global__ __launch_bounds__(256) void gemm_wmma_k(
    const float* __restrict__ A, const float* __restrict__ W,
    const float* __restrict__ bias, const float* __restrict__ res,
    float* __restrict__ C, int M, int N, int K, int act)
{
  __shared__ unsigned short As[BM][BK + 2];
  __shared__ unsigned short Ws[BN][BK + 2];
  const int bm = blockIdx.x * BM, bn = blockIdx.y * BN;
  const int tid = threadIdx.x, wave = tid >> 5, lane = tid & 31;
  const int wm = (wave & 3) * 16, wn = (wave >> 2) * 32;
  const int row = lane & 15, grp = lane >> 4;
  const bool interiorMN = (bm + BM <= M) && (bn + BN <= N);
  v8f acc0 = {0.f,0.f,0.f,0.f,0.f,0.f,0.f,0.f};
  v8f acc1 = {0.f,0.f,0.f,0.f,0.f,0.f,0.f,0.f};

#pragma unroll 1
  for (int k0 = 0; k0 < K; k0 += BK) {
    // speculative prefetch of next K slab (global_prefetch_b8)
    int pk = k0 + BK;
    if (pk < K) {
      int r = tid & 63;
      int gm = bm + r; gm = gm < M ? gm : M - 1;
      int gn = bn + r; gn = gn < N ? gn : N - 1;
      __builtin_prefetch(&A[(size_t)gm * K + pk], 0, 1);
      __builtin_prefetch(&W[(size_t)gn * K + pk], 0, 1);
    }
    if (interiorMN && (k0 + BK <= K)) {
      // fast path: fully in-range, vectorized b128 loads, no predicates
#pragma unroll
      for (int i = tid; i < (BM * BK) / 4; i += 256) {
        int r = i >> 3, c = (i & 7) * 4;
        float4 va = *(const float4*)&A[(size_t)(bm + r) * K + k0 + c];
        As[r][c + 0] = f2bf(va.x); As[r][c + 1] = f2bf(va.y);
        As[r][c + 2] = f2bf(va.z); As[r][c + 3] = f2bf(va.w);
        float4 vw = *(const float4*)&W[(size_t)(bn + r) * K + k0 + c];
        Ws[r][c + 0] = f2bf(vw.x); Ws[r][c + 1] = f2bf(vw.y);
        Ws[r][c + 2] = f2bf(vw.z); Ws[r][c + 3] = f2bf(vw.w);
      }
    } else {
      // boundary path: guarded scalar loads
#pragma unroll
      for (int i = tid; i < BM * BK; i += 256) {
        int r = i >> 5, c = i & 31;
        int gm = bm + r, gk = k0 + c;
        unsigned short ha = 0, hw = 0;
        if ((gm < M) && (gk < K)) ha = f2bf(A[(size_t)gm * K + gk]);
        int gn = bn + r;
        if ((gn < N) && (gk < K)) hw = f2bf(W[(size_t)gn * K + gk]);
        As[r][c] = ha;
        Ws[r][c] = hw;
      }
    }
    __syncthreads();
    union { v16bf v; unsigned short s[16]; } ua, ub0, ub1;
#pragma unroll
    for (int e = 0; e < 16; e++) {
      int v = e >> 1, h = e & 1;
      int ka = ((v & 3) * 2 + h) + (grp << 3) + ((v >> 2) << 4); // A 16x32 layout (ISA 7.12.2)
      int kb = ((v & 3) * 2 + h) + (grp << 3) + ((v >> 2) << 4); // B 32x16 layout
      ua.s[e]  = As[wm + row][ka];
      ub0.s[e] = Ws[wn + row][kb];
      ub1.s[e] = Ws[wn + 16 + row][kb];
    }
    acc0 = __builtin_amdgcn_wmma_f32_16x16x32_bf16(false, ua.v, false, ub0.v,
                                                   (short)0, acc0, false, false);
    acc1 = __builtin_amdgcn_wmma_f32_16x16x32_bf16(false, ua.v, false, ub1.v,
                                                   (short)0, acc1, false, false);
    __syncthreads();
  }
  int n0 = bn + wn + row;
  int n1 = n0 + 16;
#pragma unroll
  for (int v = 0; v < 8; v++) {
    int m = bm + wm + v + grp * 8;
    if (m < M) {
      if (n0 < N) {
        float val = acc0[v];
        if (bias) val += bias[n0];
        if (act == 1) val = swishf(val);
        size_t o = (size_t)m * N + n0;
        if (res) val += res[o];
        C[o] = val;
      }
      if (n1 < N) {
        float val = acc1[v];
        if (bias) val += bias[n1];
        if (act == 1) val = swishf(val);
        size_t o = (size_t)m * N + n1;
        if (res) val += res[o];
        C[o] = val;
      }
    }
  }
}

// ---------------------------------------------------------------- rmsnorm (one block per row)
__global__ __launch_bounds__(256) void rmsnorm_k(
    const float* __restrict__ x, const float* __restrict__ gamma,
    const float* __restrict__ res, float* __restrict__ out, int d)
{
  __shared__ float buf[256];
  size_t base = (size_t)blockIdx.x * d;
  float s = 0.f;
  for (int i = threadIdx.x; i < d; i += 256) { float v = x[base + i]; s += v * v; }
  buf[threadIdx.x] = s; __syncthreads();
  for (int o = 128; o > 0; o >>= 1) {
    if ((int)threadIdx.x < o) buf[threadIdx.x] += buf[threadIdx.x + o];
    __syncthreads();
  }
  float inv = rsqrtf(buf[0] / (float)d + RMS_EPS);
  for (int i = threadIdx.x; i < d; i += 256) {
    float v = x[base + i] * inv;
    if (gamma) v *= gamma[i];
    if (res)   v += res[base + i];
    out[base + i] = v;
  }
}

// ---------------------------------------------------------------- dt = softplus(raw+bias); dA = exp(-exp(A_log)*dt)
__global__ void dt_k(const float* __restrict__ zx, int col_off, int stride,
                     const float* __restrict__ dt_bias, const float* __restrict__ A_log,
                     float* __restrict__ dt, float* __restrict__ dA, size_t rows, int H)
{
  size_t idx = (size_t)blockIdx.x * 256 + threadIdx.x;
  if (idx >= rows * (size_t)H) return;
  size_t r = idx / H; int h = (int)(idx % H);
  float v = zx[r * stride + col_off + h] + dt_bias[h];
  float sp = (v > 20.f) ? v : log1pf(__expf(v));
  dt[idx] = sp;
  dA[idx] = __expf(-__expf(A_log[h]) * sp);
}

// ---------------------------------------------------------------- depthwise causal conv (D_CONV=4) + swish
__global__ void conv_swish_k(const float* __restrict__ zx, int col_off, int stride,
                             const float* __restrict__ cw, const float* __restrict__ cb,
                             float* __restrict__ out, int L, int cch, size_t total)
{
  size_t idx = (size_t)blockIdx.x * 256 + threadIdx.x;
  if (idx >= total) return;
  int c = (int)(idx % cch);
  size_t bl = idx / cch;
  int l = (int)(bl % L);
  size_t b = bl / L;
  float acc = cb[c];
#pragma unroll
  for (int k = 0; k < DCONV; k++) {
    int ls = l + k - (DCONV - 1);
    if (ls >= 0) acc += zx[((size_t)(b * L + ls)) * stride + col_off + c] * cw[c * DCONV + k];
  }
  out[idx] = swishf(acc);
}

// ---------------------------------------------------------------- selective scan: 1 block per (b,h)
// thread p owns state row h[p, 0..63] in VGPRs; B/C staged in LDS per step.
__global__ void scan_k(const float* __restrict__ xc, const float* __restrict__ dt,
                       const float* __restrict__ dA, const float* __restrict__ Dv,
                       float* __restrict__ y, int L, int H, int hd, int cch)
{
  int b = blockIdx.x / H, h = blockIdx.x % H, p = threadIdx.x;
  int di = H * hd;
  float hst[DSTATE];
#pragma unroll
  for (int n = 0; n < DSTATE; n++) hst[n] = 0.f;
  __shared__ float Bs[DSTATE], Cs[DSTATE];
  for (int t = 0; t < L; t++) {
    size_t rb = (size_t)b * L + t;
    for (int i = p; i < 2 * DSTATE; i += hd) {
      float v = xc[rb * cch + di + i];
      if (i < DSTATE) Bs[i] = v; else Cs[i - DSTATE] = v;
    }
    __syncthreads();
    float dtv = dt[rb * H + h], dav = dA[rb * H + h];
    float xv = xc[rb * cch + h * hd + p];
    float dx = dtv * xv, acc = 0.f;
#pragma unroll
    for (int n = 0; n < DSTATE; n++) { hst[n] = hst[n] * dav + dx * Bs[n]; acc += hst[n] * Cs[n]; }
    y[rb * (size_t)di + h * hd + p] = acc + Dv[h] * xv;
    __syncthreads();
  }
}

// ---------------------------------------------------------------- elementwise helpers
__global__ void gate_k(const float* __restrict__ y, const float* __restrict__ zx,
                       int zstride, float* __restrict__ g, size_t total, int di)
{
  size_t idx = (size_t)blockIdx.x * 256 + threadIdx.x;
  if (idx >= total) return;
  size_t r = idx / di; int i = (int)(idx % di);
  g[idx] = y[idx] * swishf(zx[r * zstride + i]);   // z occupies cols [0,di)
}
__global__ void rev_k(const float* __restrict__ in, float* __restrict__ out,
                      int L, int d, size_t total)
{
  size_t idx = (size_t)blockIdx.x * 256 + threadIdx.x;
  if (idx >= total) return;
  int i = (int)(idx % d);
  size_t bl = idx / d;
  int l = (int)(bl % L);
  size_t b = bl / L;
  out[((b * L + (size_t)(L - 1 - l)) * d) + i] = in[idx];
}
__global__ void mul_k(const float* a, const float* b, float* o, size_t total) {
  size_t i = (size_t)blockIdx.x * 256 + threadIdx.x;
  if (i < total) o[i] = a[i] * b[i];
}
__global__ void addmul_k(const float* w, const float* f, const float* bw, float* o, size_t total) {
  size_t i = (size_t)blockIdx.x * 256 + threadIdx.x;
  if (i < total) o[i] = w[i] * (f[i] + bw[i]);
}
__global__ void add_ip_k(float* a, const float* b, size_t total) {
  size_t i = (size_t)blockIdx.x * 256 + threadIdx.x;
  if (i < total) a[i] += b[i];
}
__global__ void add2_k(float* o, const float* a, const float* b, size_t total) {
  size_t i = (size_t)blockIdx.x * 256 + threadIdx.x;
  if (i < total) o[i] = a[i] + b[i];
}
// (b,I,J) -> (b,J,I)
__global__ void transpose_k(const float* __restrict__ in, float* __restrict__ out,
                            int I, int J, size_t total)
{
  size_t idx = (size_t)blockIdx.x * 256 + threadIdx.x;
  if (idx >= total) return;
  int j = (int)(idx % J);
  size_t bi = idx / J;
  int i = (int)(bi % I);
  size_t b = bi / I;
  out[((b * J + (size_t)j) * I) + i] = in[idx];
}

// ================================================================ host side
namespace {

struct Reader { void* const* din; int idx; const float* next() { return (const float*)din[idx++]; } };

struct MambaP { const float *in_w, *conv_w, *conv_b, *dt_bias, *A_log, *D, *norm_w, *out_w; };
struct EncP {
  const float *input_w, *input_b, *weight_w, *weight_b, *fwd_w, *fwd_b;
  MambaP fwd;
  const float *bwd_w, *bwd_b;
  MambaP bwd;
  const float *out_w, *out_b;
  bool flip;
};
struct DownUp { const float *w, *b; EncP enc; };

MambaP read_mamba(Reader& r) {
  MambaP m;
  m.in_w = r.next(); m.conv_w = r.next(); m.conv_b = r.next(); m.dt_bias = r.next();
  m.A_log = r.next(); m.D = r.next(); m.norm_w = r.next(); m.out_w = r.next();
  return m;
}
EncP read_enc(Reader& r, bool flip) {
  EncP e; e.flip = flip;
  e.input_w = r.next(); e.input_b = r.next();
  e.weight_w = r.next(); e.weight_b = r.next();
  e.fwd_w = r.next(); e.fwd_b = r.next();
  e.fwd = read_mamba(r);
  if (flip) { e.bwd_w = r.next(); e.bwd_b = r.next(); e.bwd = read_mamba(r); }
  else { e.bwd_w = e.bwd_b = nullptr; }
  e.out_w = r.next(); e.out_b = r.next();
  return e;
}
DownUp read_downup(Reader& r, bool flip) {
  DownUp d; d.w = r.next(); d.b = r.next(); d.enc = read_enc(r, flip); return d;
}

struct Arena {
  char* base; size_t off, cap;
  float* alloc(size_t nfloats) {
    size_t bytes = (nfloats * sizeof(float) + 255) & ~(size_t)255;
    float* p = (float*)(base + off); off += bytes; return p;
  }
};

inline unsigned gdiv(size_t n) { return (unsigned)((n + 255) / 256); }

void gemm(hipStream_t st, const float* A, const float* W, const float* bias,
          const float* res, float* C, int M, int N, int K, int act) {
  dim3 g((unsigned)((M + BM - 1) / BM), (unsigned)((N + BN - 1) / BN));
  gemm_wmma_k<<<g, 256, 0, st>>>(A, W, bias, res, C, M, N, K, act);
}

// dest = rmsnorm(mamba2(u)) + u   (post_norm_mamba)
void mamba_post(Arena ar, hipStream_t st, const float* u, float* dest,
                const MambaP& mp, int Bb, int L, int dm, int hd) {
  int di = 2 * dm, H = di / hd, cch = di + 2 * DSTATE, Nout = 2 * di + 2 * DSTATE + H;
  int M = Bb * L;
  float* zx = ar.alloc((size_t)M * Nout);
  gemm(st, u, mp.in_w, nullptr, nullptr, zx, M, Nout, dm, 0);
  float* dt = ar.alloc((size_t)M * H);
  float* dA = ar.alloc((size_t)M * H);
  dt_k<<<gdiv((size_t)M * H), 256, 0, st>>>(zx, 2 * di + 2 * DSTATE, Nout,
                                            mp.dt_bias, mp.A_log, dt, dA, (size_t)M, H);
  float* xc = ar.alloc((size_t)M * cch);
  conv_swish_k<<<gdiv((size_t)M * cch), 256, 0, st>>>(zx, di, Nout, mp.conv_w, mp.conv_b,
                                                      xc, L, cch, (size_t)M * cch);
  float* y = ar.alloc((size_t)M * di);
  scan_k<<<Bb * H, hd, 0, st>>>(xc, dt, dA, mp.D, y, L, H, hd, cch);
  float* g = ar.alloc((size_t)M * di);
  gate_k<<<gdiv((size_t)M * di), 256, 0, st>>>(y, zx, Nout, g, (size_t)M * di, di);
  float* yn = ar.alloc((size_t)M * di);
  rmsnorm_k<<<M, 256, 0, st>>>(g, mp.norm_w, nullptr, yn, di);
  float* mo = ar.alloc((size_t)M * dm);
  gemm(st, yn, mp.out_w, nullptr, nullptr, mo, M, dm, di, 0);
  rmsnorm_k<<<M, 256, 0, st>>>(mo, nullptr, u, dest, dm);   // + residual u
}

// dest = enc_forward/enc_flip(x)
void enc_run(Arena ar, hipStream_t st, const float* x, float* dest,
             const EncP& p, int Bb, int L, int d, int hd) {
  int M = Bb * L, d2 = 2 * d;
  size_t tot = (size_t)M * d2;
  float* xn = ar.alloc((size_t)M * d);
  rmsnorm_k<<<M, 256, 0, st>>>(x, nullptr, nullptr, xn, d);
  float* sb = ar.alloc(tot);
  gemm(st, xn, p.input_w, p.input_b, nullptr, sb, M, d2, d, 0);
  float* fb = ar.alloc(tot);
  gemm(st, sb, p.fwd_w, p.fwd_b, nullptr, fb, M, d2, d2, 0);
  float* fpm = ar.alloc(tot);
  mamba_post(ar, st, fb, fpm, p.fwd, Bb, L, d2, hd);
  float* wb = ar.alloc(tot);
  gemm(st, xn, p.weight_w, p.weight_b, nullptr, wb, M, d2, d, 1 /*swish*/);
  float* t = ar.alloc(tot);
  if (p.flip) {
    float* srev = ar.alloc(tot);
    rev_k<<<gdiv(tot), 256, 0, st>>>(sb, srev, L, d2, tot);
    float* bb = ar.alloc(tot);
    gemm(st, srev, p.bwd_w, p.bwd_b, nullptr, bb, M, d2, d2, 0);
    float* bpm = ar.alloc(tot);
    mamba_post(ar, st, bb, bpm, p.bwd, Bb, L, d2, hd);
    float* brev = ar.alloc(tot);
    rev_k<<<gdiv(tot), 256, 0, st>>>(bpm, brev, L, d2, tot);
    addmul_k<<<gdiv(tot), 256, 0, st>>>(wb, fpm, brev, t, tot);
  } else {
    mul_k<<<gdiv(tot), 256, 0, st>>>(wb, fpm, t, tot);
  }
  gemm(st, t, p.out_w, p.out_b, x /*residual*/, dest, M, d, d2, 0);
}

} // namespace

extern "C" void kernel_launch(void* const* d_in, const int* in_sizes, int n_in,
                              void* d_out, int out_size, void* d_ws, size_t ws_size,
                              hipStream_t stream) {
  (void)in_sizes; (void)out_size; (void)ws_size;
  if (n_in < 299) return;   // expected flattened leaf count

  Reader r{d_in, 0};
  const float* x_in = r.next();

  // ---- params in setup_inputs() insertion order: urms then urmc
  DownUp dS[2]; for (int i = 0; i < 2; i++) dS[i] = read_downup(r, true);
  EncP  midS = read_enc(r, true);
  DownUp uS[2]; for (int i = 0; i < 2; i++) uS[i] = read_downup(r, true);
  DownUp dC[4]; for (int i = 0; i < 4; i++) dC[i] = read_downup(r, false);
  EncP  midC = read_enc(r, false);
  DownUp uC[4]; for (int i = 0; i < 4; i++) uC[i] = read_downup(r, false);

  const int Bb = 8, C = 256, L0 = 512;
  const size_t TOT = (size_t)Bb * C * L0;

  Arena ar{(char*)d_ws, 0, ws_size};
  float* CUR = ar.alloc(TOT);
  float* TMP = ar.alloc(TOT);
  float* NXT = ar.alloc(TOT);
  float* Rs[2]; Rs[0] = ar.alloc((size_t)Bb * C * 512); Rs[1] = ar.alloc((size_t)Bb * C * 256);
  float* Rc[4];
  { int cc = 256; for (int i = 0; i < 4; i++) { Rc[i] = ar.alloc((size_t)Bb * 512 * cc); cc /= 2; } }

  (void)hipMemcpyAsync(CUR, x_in, TOT * sizeof(float), hipMemcpyDeviceToDevice, stream);

  // ================= urms: sequence-length U-Net, tokens = channels (256), hd=64
  int Lc = L0;
  for (int lvl = 0; lvl < 2; lvl++) {
    (void)hipMemcpyAsync(Rs[lvl], CUR, (size_t)Bb * C * Lc * sizeof(float),
                         hipMemcpyDeviceToDevice, stream);
    gemm(stream, CUR, dS[lvl].w, dS[lvl].b, nullptr, TMP, Bb * C, Lc / 2, Lc, 0);
    enc_run(ar, stream, TMP, NXT, dS[lvl].enc, Bb, C, Lc / 2, 64);
    float* t = CUR; CUR = NXT; NXT = t;
    Lc /= 2;
  }
  enc_run(ar, stream, CUR, NXT, midS, Bb, C, Lc, 64);
  { float* t = CUR; CUR = NXT; NXT = t; }
  for (int lvl = 0; lvl < 2; lvl++) {
    gemm(stream, CUR, uS[lvl].w, uS[lvl].b, nullptr, TMP, Bb * C, 2 * Lc, Lc, 0);
    enc_run(ar, stream, TMP, NXT, uS[lvl].enc, Bb, C, 2 * Lc, 64);
    float* t = CUR; CUR = NXT; NXT = t;
    Lc *= 2;
    size_t n = (size_t)Bb * C * Lc;
    add_ip_k<<<gdiv(n), 256, 0, stream>>>(CUR, Rs[1 - lvl], n);
  }

  // ================= urmc: channel U-Net, tokens = seq (512), hd=32; work in (b,l,c)
  transpose_k<<<gdiv(TOT), 256, 0, stream>>>(CUR, TMP, C, L0, TOT); // (b,c,l)->(b,l,c)
  { float* t = CUR; CUR = TMP; TMP = t; }
  int Cc = 256;
  for (int lvl = 0; lvl < 4; lvl++) {
    (void)hipMemcpyAsync(Rc[lvl], CUR, (size_t)Bb * L0 * Cc * sizeof(float),
                         hipMemcpyDeviceToDevice, stream);
    gemm(stream, CUR, dC[lvl].w, dC[lvl].b, nullptr, TMP, Bb * L0, Cc / 2, Cc, 0);
    enc_run(ar, stream, TMP, NXT, dC[lvl].enc, Bb, L0, Cc / 2, 32);
    float* t = CUR; CUR = NXT; NXT = t;
    Cc /= 2;
  }
  enc_run(ar, stream, CUR, NXT, midC, Bb, L0, Cc, 32);
  { float* t = CUR; CUR = NXT; NXT = t; }
  for (int lvl = 0; lvl < 4; lvl++) {
    gemm(stream, CUR, uC[lvl].w, uC[lvl].b, nullptr, TMP, Bb * L0, 2 * Cc, Cc, 0);
    enc_run(ar, stream, TMP, NXT, uC[lvl].enc, Bb, L0, 2 * Cc, 32);
    float* t = CUR; CUR = NXT; NXT = t;
    Cc *= 2;
    size_t n = (size_t)Bb * L0 * Cc;
    add_ip_k<<<gdiv(n), 256, 0, stream>>>(CUR, Rc[3 - lvl], n);
  }

  // back to (b,c,l) and final residual with original input
  transpose_k<<<gdiv(TOT), 256, 0, stream>>>(CUR, TMP, L0, C, TOT); // (b,l,c)->(b,c,l)
  add2_k<<<gdiv(TOT), 256, 0, stream>>>((float*)d_out, TMP, x_in, TOT);
}
